// Seq2seqAtt_936302870892
// MI455X (gfx1250) — matrix operand order repaired and checked by reading, not compiled, not run on hardware
//
#include <hip/hip_runtime.h>
#include <hip/hip_bf16.h>

// ---------------------------------------------------------------------------
// Seq2seq + Luong attention for MI455X (gfx1250), bf16 WMMA everywhere.
//   V = H = 2048, B = 64, Ti = 128, To = 129 (padded to 144 for 16-tiles).
// All matrix math via v_wmma_f32_16x16x32_bf16 (f32 accumulate).
// Big GEMMs (energy, preds) use a 64x16-per-wave register-blocked kernel:
// 4 WMMAs share one B fragment -> 4x less L2 weight traffic.
// ---------------------------------------------------------------------------

typedef __hip_bfloat16 bf16;
typedef __attribute__((ext_vector_type(16))) __bf16 bf16x16;
typedef __attribute__((ext_vector_type(8)))  float  f32x8;

constexpr int kV   = 2048;
constexpr int kH   = 2048;
constexpr int kB   = 64;
constexpr int kTI  = 128;
constexpr int kTO  = 129;
constexpr int kTOP = 144;          // To padded to multiple of 16

struct Frag256 { uint4 lo; uint4 hi; };   // 32 bytes == 16 bf16

// A-operand (16x32, M x K): lane half=lane>>4, row=lane&15.
// Holds A[row][k0+half*16 .. +15] (v[0..15]) -- fully contiguous.
__device__ __forceinline__ bf16x16 load_a_frag(const bf16* p) {
  Frag256 f;
  f.lo = *(const uint4*)(p);        // 8 bf16
  f.hi = *(const uint4*)(p + 8);    // +8 elements
  return __builtin_bit_cast(bf16x16, f);
}
// B-operand (32x16, K x N) from Bt stored [N,K] row-major:
// lane (half,col) holds Bt[col][k0+half*16 .. +15] -- fully contiguous.
__device__ __forceinline__ bf16x16 load_b_frag(const bf16* p) {
  Frag256 f;
  f.lo = *(const uint4*)(p);
  f.hi = *(const uint4*)(p + 8);
  return __builtin_bit_cast(bf16x16, f);
}

__device__ __forceinline__ f32x8 wmma_bf16(bf16x16 a, bf16x16 b, f32x8 c) {
  return __builtin_amdgcn_wmma_f32_16x16x32_bf16(
      /*neg_a=*/false, a, /*neg_b=*/false, b,
      /*c_mod=*/(short)0, c, /*reuse_a=*/false, /*reuse_b=*/false);
}

// ---------------------------------------------------------------------------
// Generic batched GEMM (1 tile / wave):
//   C[z][m][n] = sum_k A[z][m][k] * Bt[z][n][k] (+bias[n])
// Output is f32 (Cf) or bf16 (Cb) -- exactly one non-null. m >= Mstore skipped.
// Block = 256 threads = 8 waves; wave w owns tile (m0, n0 = (bx*8+w)*16).
// ---------------------------------------------------------------------------
__global__ __launch_bounds__(256) void gemm_wmma_bf16(
    const bf16* __restrict__ A,  long lda, long sA,
    const bf16* __restrict__ Bt, long ldb, long sB,
    float* __restrict__ Cf, bf16* __restrict__ Cb, long ldc, long sC,
    const float* __restrict__ bias, int N, int K, int Mstore)
{
  const int wave = threadIdx.x >> 5;
  const int lane = threadIdx.x & 31;
  const int half = lane >> 4;
  const int lc   = lane & 15;
  const int n0   = (blockIdx.x * 8 + wave) * 16;
  const int m0   = blockIdx.y * 16;
  if (n0 >= N) return;
  const long z = blockIdx.z;

  const bf16* Ap = A  + z * sA + (long)(m0 + lc) * lda + half * 16;
  const bf16* Bp = Bt + z * sB + (long)(n0 + lc) * ldb + half * 16;

  f32x8 acc = {};
  for (int k0 = 0; k0 < K; k0 += 32) {
    acc = wmma_bf16(load_a_frag(Ap + k0), load_b_frag(Bp + k0), acc);
  }

  const float bv = bias ? bias[n0 + lc] : 0.0f;
  const long cbase = z * sC + (long)(n0 + lc);
  #pragma unroll
  for (int r = 0; r < 8; ++r) {
    const int m = m0 + half * 8 + r;
    if (m >= Mstore) continue;
    const float v = acc[r] + bv;
    if (Cf) Cf[cbase + (long)m * ldc] = v;
    else    Cb[cbase + (long)m * ldc] = __float2bfloat16(v);
  }
}

// ---------------------------------------------------------------------------
// Register-blocked GEMM (64x16 per wave, 4 m-tiles share one B fragment).
// Requires M % 64 == 0. Unbatched. C = A[M,K] * Bt[N,K] (+bias[n]).
// ---------------------------------------------------------------------------
__global__ __launch_bounds__(256) void gemm_wmma_bf16_m4(
    const bf16* __restrict__ A,  long lda,
    const bf16* __restrict__ Bt, long ldb,
    float* __restrict__ Cf, bf16* __restrict__ Cb, long ldc,
    const float* __restrict__ bias, int N, int K)
{
  const int wave = threadIdx.x >> 5;
  const int lane = threadIdx.x & 31;
  const int half = lane >> 4;
  const int lc   = lane & 15;
  const int n0   = (blockIdx.x * 8 + wave) * 16;
  const int m0   = blockIdx.y * 64;
  if (n0 >= N) return;

  const bf16* Ap = A  + (long)(m0 + lc) * lda + half * 16;
  const bf16* Bp = Bt + (long)(n0 + lc) * ldb + half * 16;
  const long a16 = 16 * lda;

  f32x8 acc0 = {}, acc1 = {}, acc2 = {}, acc3 = {};
  for (int k0 = 0; k0 < K; k0 += 32) {
    const bf16x16 b = load_b_frag(Bp + k0);
    acc0 = wmma_bf16(load_a_frag(Ap + k0),            b, acc0);
    acc1 = wmma_bf16(load_a_frag(Ap + k0 + a16),      b, acc1);
    acc2 = wmma_bf16(load_a_frag(Ap + k0 + 2 * a16),  b, acc2);
    acc3 = wmma_bf16(load_a_frag(Ap + k0 + 3 * a16),  b, acc3);
  }

  const float bv = bias ? bias[n0 + lc] : 0.0f;
  float* cf = Cf ? Cf + (long)(m0 + half * 8) * ldc + (n0 + lc) : nullptr;
  bf16*  cb = Cb ? Cb + (long)(m0 + half * 8) * ldc + (n0 + lc) : nullptr;
  const long c16 = 16 * ldc;
  f32x8 accs[4] = {acc0, acc1, acc2, acc3};
  #pragma unroll
  for (int mt = 0; mt < 4; ++mt) {
    #pragma unroll
    for (int r = 0; r < 8; ++r) {
      const float v = accs[mt][r] + bv;
      if (Cf) cf[mt * c16 + (long)r * ldc] = v;
      else    cb[mt * c16 + (long)r * ldc] = __float2bfloat16(v);
    }
  }
}

// ---------------------------------------------------------------------------
// One RNN time step:  hout[b][n] = tanh( Wih[n][idx[b]] + bih[n] + bhh[n]
//                                        + sum_k hprev[b][k] * Whh[n][k] )
// M = kB (64), N = K = kH (2048). One-hot input projection folded into the
// epilogue as an embedding gather (Wih stays f32, resident in L2).
// hout2 (optional) mirrors the store into the feats concat buffer.
// ---------------------------------------------------------------------------
__global__ __launch_bounds__(256) void rnn_step_wmma(
    const bf16* __restrict__ hprev, const bf16* __restrict__ Whh,
    const float* __restrict__ Wih, const float* __restrict__ bih,
    const float* __restrict__ bhh, const int* __restrict__ idx,
    bf16* __restrict__ hout, bf16* __restrict__ hout2, long ld2)
{
  const int wave = threadIdx.x >> 5;
  const int lane = threadIdx.x & 31;
  const int half = lane >> 4;
  const int lc   = lane & 15;
  const int n0   = (blockIdx.x * 8 + wave) * 16;
  const int m0   = blockIdx.y * 16;

  const bf16* Ap = hprev + (long)(m0 + lc) * kH + half * 16;
  const bf16* Bp = Whh   + (long)(n0 + lc) * kH + half * 16;

  f32x8 acc = {};
  for (int k0 = 0; k0 < kH; k0 += 32) {
    acc = wmma_bf16(load_a_frag(Ap + k0), load_b_frag(Bp + k0), acc);
  }

  const int n = n0 + lc;
  const float bsum = bih[n] + bhh[n];
  const float* wrow = Wih + (size_t)n * kV;
  #pragma unroll
  for (int r = 0; r < 8; ++r) {
    const int b = m0 + half * 8 + r;            // batch row (m)
    const float xp = wrow[idx[b]];              // one-hot input projection
    const float v  = tanhf(acc[r] + xp + bsum);
    const bf16 hv  = __float2bfloat16(v);
    hout[(size_t)b * kH + n] = hv;
    if (hout2) hout2[(size_t)b * ld2 + n] = hv;
  }
}

// --------------------------- helper kernels --------------------------------

__global__ void cvt_f32_bf16(const float* __restrict__ s, bf16* __restrict__ d,
                             size_t n) {
  size_t i = blockIdx.x * (size_t)blockDim.x + threadIdx.x;
  const size_t stride = (size_t)gridDim.x * blockDim.x;
  for (; i < n; i += stride) d[i] = __float2bfloat16(s[i]);
}

__global__ void zero_fill_u32(unsigned int* __restrict__ p, size_t n) {
  size_t i = blockIdx.x * (size_t)blockDim.x + threadIdx.x;
  const size_t stride = (size_t)gridDim.x * blockDim.x;
  for (; i < n; i += stride) p[i] = 0u;
}

// enc_hs [Ti][B][H] -> enc_hsT [B][H][Ti]  (K-contiguous for context GEMM)
__global__ void transpose_enc_hs(const bf16* __restrict__ src,
                                 bf16* __restrict__ dst) {
  size_t t = blockIdx.x * (size_t)blockDim.x + threadIdx.x;
  const size_t total = (size_t)kTI * kB * kH;
  if (t >= total) return;
  const int i = (int)(t / (kB * kH));
  const int r = (int)(t % (kB * kH));
  const int b = r / kH;
  const int h = r % kH;
  dst[((size_t)b * kH + h) * kTI + i] = src[t];
}

// softmax over Ti for one (b, o) row: scores f32 [B*TOP][Ti] -> att bf16
__global__ void att_softmax(const float* __restrict__ scores,
                            bf16* __restrict__ att) {
  const int row = blockIdx.x;
  const int i   = threadIdx.x;                 // Ti = 128 threads
  const float v = scores[(size_t)row * kTI + i];
  __shared__ float red[kTI];
  red[i] = v; __syncthreads();
  for (int s = kTI / 2; s > 0; s >>= 1) {
    if (i < s) red[i] = fmaxf(red[i], red[i + s]);
    __syncthreads();
  }
  const float m = red[0]; __syncthreads();
  const float e = __expf(v - m);
  red[i] = e; __syncthreads();
  for (int s = kTI / 2; s > 0; s >>= 1) {
    if (i < s) red[i] += red[i + s];
    __syncthreads();
  }
  att[(size_t)row * kTI + i] = __float2bfloat16(e / red[0]);
}

// in-place log_softmax over V per row of d_out [To*B][V]
__global__ void log_softmax_rows(float* __restrict__ x) {
  float* p = x + (size_t)blockIdx.x * kV;
  const int tid = threadIdx.x;                 // 256 threads
  __shared__ float red[256];
  float lm = -3.0e38f;
  for (int j = tid; j < kV; j += 256) lm = fmaxf(lm, p[j]);
  red[tid] = lm; __syncthreads();
  for (int s = 128; s > 0; s >>= 1) {
    if (tid < s) red[tid] = fmaxf(red[tid], red[tid + s]);
    __syncthreads();
  }
  const float m = red[0]; __syncthreads();
  float ls = 0.0f;
  for (int j = tid; j < kV; j += 256) ls += __expf(p[j] - m);
  red[tid] = ls; __syncthreads();
  for (int s = 128; s > 0; s >>= 1) {
    if (tid < s) red[tid] += red[tid + s];
    __syncthreads();
  }
  const float lse = m + __logf(red[0]);
  __syncthreads();
  for (int j = tid; j < kV; j += 256) p[j] -= lse;
}

// ---------------------------------------------------------------------------

extern "C" void kernel_launch(void* const* d_in, const int* in_sizes, int n_in,
                              void* d_out, int out_size, void* d_ws, size_t ws_size,
                              hipStream_t stream) {
  (void)in_sizes; (void)n_in; (void)out_size; (void)ws_size;

  const int*   inp      = (const int*)  d_in[0];   // [Ti,B]
  const int*   out_seq  = (const int*)  d_in[1];   // [To,B]
  const float* enc_Wih  = (const float*)d_in[2];   // [H,V]
  const float* enc_Whh  = (const float*)d_in[3];   // [H,H]
  const float* enc_bih  = (const float*)d_in[4];
  const float* enc_bhh  = (const float*)d_in[5];
  const float* dec_Wih  = (const float*)d_in[6];
  const float* dec_Whh  = (const float*)d_in[7];
  const float* dec_bih  = (const float*)d_in[8];
  const float* dec_bhh  = (const float*)d_in[9];
  const float* W_att    = (const float*)d_in[10];  // [H,H]
  const float* out_W    = (const float*)d_in[11];  // [V,2H]
  const float* out_b    = (const float*)d_in[12];  // [V]
  float* out = (float*)d_out;                      // [To,B,V] f32

  // ---- carve workspace (~256 MB) ----
  char* ws = (char*)d_ws;
  size_t off = 0;
  auto carve = [&](size_t bytes) -> char* {
    char* p = ws + off;
    off = (off + bytes + 255) & ~(size_t)255;
    return p;
  };
  bf16*  encWhh_b = (bf16*) carve((size_t)kH * kH * 2);
  bf16*  decWhh_b = (bf16*) carve((size_t)kH * kH * 2);
  bf16*  Watt_b   = (bf16*) carve((size_t)kH * kH * 2);
  bf16*  outW_b   = (bf16*) carve((size_t)kV * 2 * kH * 2);
  bf16*  h0       = (bf16*) carve((size_t)kB * kH * 2);
  bf16*  enc_hs   = (bf16*) carve((size_t)kTI * kB * kH * 2);   // [Ti][B][H]
  bf16*  dec_hs   = (bf16*) carve((size_t)kTOP * kB * kH * 2);  // [TOP][B][H]
  bf16*  energy   = (bf16*) carve((size_t)kTI * kB * kH * 2);   // [Ti][B][H]
  bf16*  enc_hsT  = (bf16*) carve((size_t)kB * kH * kTI * 2);   // [B][H][Ti]
  float* scores   = (float*)carve((size_t)kB * kTOP * kTI * 4); // [B][TOP][Ti]
  bf16*  att_b    = (bf16*) carve((size_t)kB * kTOP * kTI * 2); // [B][TOP][Ti]
  bf16*  feats    = (bf16*) carve((size_t)kTO * kB * 2 * kH * 2);// [To*B][2H]

  // ---- init: zero h0 and dec_hs (padding rows 129..143 must be zero) ----
  zero_fill_u32<<<256, 256, 0, stream>>>((unsigned int*)h0,
                                         (size_t)kB * kH * 2 / 4);
  zero_fill_u32<<<4096, 256, 0, stream>>>((unsigned int*)dec_hs,
                                          (size_t)kTOP * kB * kH * 2 / 4);

  // ---- convert weights f32 -> bf16 ----
  cvt_f32_bf16<<<4096, 256, 0, stream>>>(enc_Whh, encWhh_b, (size_t)kH * kH);
  cvt_f32_bf16<<<4096, 256, 0, stream>>>(dec_Whh, decWhh_b, (size_t)kH * kH);
  cvt_f32_bf16<<<4096, 256, 0, stream>>>(W_att,   Watt_b,   (size_t)kH * kH);
  cvt_f32_bf16<<<4096, 256, 0, stream>>>(out_W,   outW_b,   (size_t)kV * 2 * kH);

  const dim3 rnnGrid(kH / 128, kB / 16, 1);   // 16 x 4 blocks, 8 waves each

  // ---- encoder recurrence ----
  for (int t = 0; t < kTI; ++t) {
    const bf16* hprev = (t == 0) ? h0 : enc_hs + (size_t)(t - 1) * kB * kH;
    rnn_step_wmma<<<rnnGrid, 256, 0, stream>>>(
        hprev, encWhh_b, enc_Wih, enc_bih, enc_bhh, inp + (size_t)t * kB,
        enc_hs + (size_t)t * kB * kH, (bf16*)nullptr, 0);
  }

  // ---- decoder recurrence (h0 = enc_hT); mirror store into feats[:, 0:H] ----
  for (int t = 0; t < kTO; ++t) {
    const bf16* hprev = (t == 0) ? enc_hs + (size_t)(kTI - 1) * kB * kH
                                 : dec_hs + (size_t)(t - 1) * kB * kH;
    rnn_step_wmma<<<rnnGrid, 256, 0, stream>>>(
        hprev, decWhh_b, dec_Wih, dec_bih, dec_bhh, out_seq + (size_t)t * kB,
        dec_hs + (size_t)t * kB * kH,
        feats + (size_t)t * kB * 2 * kH, (long)(2 * kH));
  }

  // ---- energy[i,b,n] = sum_h enc_hs[i,b,h] * W_att[n,h]  (M=Ti*B=8192) ----
  gemm_wmma_bf16_m4<<<dim3(kH / 128, (kTI * kB) / 64, 1), 256, 0, stream>>>(
      enc_hs, kH, Watt_b, kH,
      nullptr, energy, kH, nullptr, kH, kH);

  // ---- enc_hs -> enc_hsT for context GEMM ----
  transpose_enc_hs<<<((size_t)kTI * kB * kH + 255) / 256, 256, 0, stream>>>(
      enc_hs, enc_hsT);

  // ---- scores[b][o][i] = sum_h dec_hs[o,b,h] * energy[i,b,h] (batched) ----
  gemm_wmma_bf16<<<dim3(1, kTOP / 16, kB), 256, 0, stream>>>(
      dec_hs, (long)kB * kH, kH,            // A: rows o, batch b
      energy, (long)kB * kH, kH,            // Bt: rows i, batch b
      scores, (bf16*)nullptr, kTI, (long)kTOP * kTI,
      nullptr, kTI, kH, kTOP);

  // ---- softmax over Ti ----
  att_softmax<<<kB * kTOP, kTI, 0, stream>>>(scores, att_b);

  // ---- context[o,b,h] = sum_i att[b][o][i] * enc_hsT[b][h][i]
  //      stored straight into feats[:, H:2H] ----
  gemm_wmma_bf16<<<dim3(kH / 128, kTOP / 16, kB), 256, 0, stream>>>(
      att_b,   kTI, (long)kTOP * kTI,
      enc_hsT, kTI, (long)kH * kTI,
      nullptr, feats + kH, (long)kB * 2 * kH, (long)2 * kH,
      nullptr, kH, kTI, kTO /*skip padded o rows*/);

  // ---- preds = feats @ out_W^T + out_b -> d_out (f32); M = To*B = 8256 ----
  gemm_wmma_bf16_m4<<<dim3(kV / 128, (kTO * kB) / 64, 1), 256, 0, stream>>>(
      feats, (long)2 * kH, outW_b, (long)2 * kH,
      out, nullptr, kV, out_b, kV, 2 * kH);

  // ---- in-place log_softmax over V ----
  log_softmax_rows<<<kTO * kB, 256, 0, stream>>>(out);
}
